// CasualSelfAttention_13632226198100
// MI455X (gfx1250) — matrix operand order repaired and checked by reading, not compiled, not run on hardware
//
#include <hip/hip_runtime.h>
#include <hip/hip_bf16.h>

// ---------- CDNA5 WMMA / TDM types ----------
typedef __attribute__((ext_vector_type(16))) _Float16 v16h;
typedef __attribute__((ext_vector_type(8)))  _Float16 v8h;
typedef __attribute__((ext_vector_type(4)))  _Float16 v4h;
typedef __attribute__((ext_vector_type(8)))  float    v8f;
typedef __attribute__((ext_vector_type(4)))  unsigned u32x4;
typedef __attribute__((ext_vector_type(8)))  int      i32x8;
typedef __attribute__((ext_vector_type(4)))  int      i32x4;

#define WMMA_F16(A, B, C) \
  __builtin_amdgcn_wmma_f32_16x16x32_f16(false, (A), false, (B), (short)0, (C), false, false)

static constexpr int Bv_ = 4, S_ = 2048, E_ = 1024, H_ = 16, D_ = 64;

// Compose a v16h A/B fragment from two 16-byte LDS/global chunks.
__device__ __forceinline__ v16h ld2x8(const _Float16* lo, const _Float16* hi) {
  v8h a = *(const v8h*)lo;
  v8h b = *(const v8h*)hi;
  return __builtin_shufflevector(a, b, 0,1,2,3,4,5,6,7,8,9,10,11,12,13,14,15);
}

// LDS byte offset of a __shared__ object: generic pointers to LDS carry the
// workgroup-relative LDS address in their low 32 bits (flat LDS aperture).
__device__ __forceinline__ unsigned lds_off(const void* p) {
  return (unsigned)(unsigned long long)p;
}

// Issue one 2-D TDM tile load (f16 elements, row padding configured by w0).
//   lds     : LDS byte address of tile base
//   ga      : global byte address of tile start
//   w0      : D# group1 word0 (data_size | pad_enable | pad_interval | pad_amount)
//   dim0/1  : tensor dims (elements / rows)
//   til0/1  : tile dims (elements / rows)
//   strd    : tensor dim0 stride (elements)
__device__ __forceinline__ void tdm_load_2d(unsigned lds, unsigned long long ga,
                                            int w0, int dim0, int dim1,
                                            int til0, int til1, int strd) {
  u32x4 g0;
  g0[0] = 1u;                                    // count=1 (valid user descriptor)
  g0[1] = lds;                                   // lds_addr [63:32]
  g0[2] = (unsigned)ga;                          // global_addr [95:64]
  g0[3] = (unsigned)(ga >> 32) | 0x80000000u;    // global_addr hi + type=2 ("image")
  i32x8 g1;
  g1[0] = w0;                                    // wg_mask=0 | data_size | pad ctrl
  g1[1] = (dim0 & 0xffff) << 16;                 // tensor_dim0 [79:48]
  g1[2] = ((dim0 >> 16) & 0xffff) | ((dim1 & 0xffff) << 16);   // dim0 hi | dim1 lo
  g1[3] = ((dim1 >> 16) & 0xffff) | (til0 << 16);              // dim1 hi | tile_dim0
  g1[4] = til1 & 0xffff;                         // tile_dim1 (tile_dim2 = 0)
  g1[5] = strd;                                  // tensor_dim0_stride lo32
  g1[6] = 0;
  g1[7] = 0;                                     // dim1_stride unused for 2-D tiles
  i32x4 z4 = {0, 0, 0, 0};                       // groups 2/3: higher dims unused
  i32x8 z8 = {0, 0, 0, 0, 0, 0, 0, 0};
  // This toolchain exposes the 6-arg (clang-23 / therock-10.0) builtin form.
  __builtin_amdgcn_tensor_load_to_lds(g0, g1, z4, z4, z8, 0);
}

// D# group1 word0 values: data_size=1 (2B), pad_enable, pad_interval, pad_amount
//  K tile rows: 64 halves = 32 DWORDs data + 4 DWORDs pad -> interval code 4, amount code 3
//  V^T tile rows: 32 halves = 16 DWORDs data + 4 DWORDs pad -> interval code 3, amount code 3
static constexpr int TDM_W0_K = (1 << 16) | (1 << 20) | (4 << 22) | (3 << 25);
static constexpr int TDM_W0_V = (1 << 16) | (1 << 20) | (3 << 22) | (3 << 25);

// =====================================================================
// Kernel 1: fused QKV projection.  out = x @ W^T  (M=B*S=8192, N=K=1024)
// Q -> f16 [B,H,S,D] pre-scaled by 1/sqrt(D); K -> f16 [B,H,S,D];
// V -> f16 TRANSPOSED [B,H,D,S].
// =====================================================================
__global__ __launch_bounds__(256) void qkv_proj_kernel(
    const float* __restrict__ x,
    const float* __restrict__ Wq, const float* __restrict__ Wk,
    const float* __restrict__ Wv,
    _Float16* __restrict__ Qh, _Float16* __restrict__ Kh,
    _Float16* __restrict__ VT)
{
  __shared__ _Float16 aT[128][40];
  __shared__ _Float16 wT[64][40];

  const int t  = threadIdx.x;
  const int wv = t >> 5;
  const int l  = t & 31;
  const int lm = l & 15;
  const bool hihalf = (l >= 16);
  const int sel8  = hihalf ? 8  : 0;
  const int sel16 = hihalf ? 16 : 0;

  const int ntile = blockIdx.x * 64;
  const int mtile = blockIdx.y * 128;
  const int w     = blockIdx.z;                 // 0:Q 1:K 2:V
  const float* W  = (w == 0) ? Wq : ((w == 1) ? Wk : Wv);

  v8f acc[4];
  for (int f = 0; f < 4; ++f) acc[f] = (v8f){0.f,0.f,0.f,0.f,0.f,0.f,0.f,0.f};

  for (int kt = 0; kt < E_ / 32; ++kt) {
    __syncthreads();
    for (int i = 0; i < 4; ++i) {               // x tile 128x32 fp32 -> f16
      int e = t + 256 * i;
      int row = e >> 3, c4 = (e & 7) * 4;
      float4 f4 = *(const float4*)(x + (size_t)(mtile + row) * E_ + kt * 32 + c4);
      v4h h; h[0]=(_Float16)f4.x; h[1]=(_Float16)f4.y; h[2]=(_Float16)f4.z; h[3]=(_Float16)f4.w;
      *(v4h*)&aT[row][c4] = h;
    }
    for (int i = 0; i < 2; ++i) {               // W tile 64x32 fp32 -> f16
      int e = t + 256 * i;
      int row = e >> 3, c4 = (e & 7) * 4;
      float4 f4 = *(const float4*)(W + (size_t)(ntile + row) * E_ + kt * 32 + c4);
      v4h h; h[0]=(_Float16)f4.x; h[1]=(_Float16)f4.y; h[2]=(_Float16)f4.z; h[3]=(_Float16)f4.w;
      *(v4h*)&wT[row][c4] = h;
    }
    if (kt + 1 < E_ / 32)
      __builtin_prefetch(x + (size_t)(mtile + (t >> 1)) * E_ + (kt + 1) * 32, 0, 3);
    __syncthreads();

    const _Float16* arow = &aT[wv * 16 + lm][0];
    v16h a = ld2x8(arow + sel16, arow + sel16 + 8);
    for (int f = 0; f < 4; ++f) {
      const _Float16* brow = &wT[f * 16 + lm][0];
      v16h bf = ld2x8(brow + sel16, brow + sel16 + 8);
      acc[f] = WMMA_F16(a, bf, acc[f]);
    }
  }

  const int mbase = mtile + wv * 16 + (hihalf ? 8 : 0);
  const int bb = mbase >> 11;
  const int s0 = mbase & (S_ - 1);
  const float scale = (w == 0) ? 0.125f : 1.0f;   // 1/sqrt(64) folded into Q
  for (int f = 0; f < 4; ++f) {
    const int n = ntile + f * 16 + lm;
    const int h = n >> 6, d = n & 63;
    if (w < 2) {
      _Float16* dst = ((w == 0) ? Qh : Kh) +
                      ((size_t)(bb * H_ + h) * S_ + s0) * D_ + d;
      for (int r = 0; r < 8; ++r) dst[(size_t)r * D_] = (_Float16)(acc[f][r] * scale);
    } else {
      v8h hv; for (int r = 0; r < 8; ++r) hv[r] = (_Float16)acc[f][r];
      *(v8h*)(VT + ((size_t)(bb * H_ + h) * D_ + d) * S_ + s0) = hv;
    }
  }
}

// =====================================================================
// Kernel 2: flash attention with TDM double-buffered K/V^T staging.
// Block = 128 query rows (8 waves x 16), grid = (S/128, B*H).
// Wave 0 drives the Tensor Data Mover; tile kt+1's DMA overlaps tile kt's
// WMMA compute; TENSORcnt is in-order so waiting <=2 = current tile ready.
// =====================================================================
__global__ __launch_bounds__(256) void attn_kernel(
    const _Float16* __restrict__ Qh, const _Float16* __restrict__ Kh,
    const _Float16* __restrict__ VT, const unsigned char* __restrict__ amask,
    _Float16* __restrict__ Oh)
{
  __shared__ _Float16 kT[2][32][72];    // double-buffered 32 keys x 64 d (+TDM pad)
  __shared__ _Float16 vT[2][64][40];    // double-buffered 64 d x 32 keys (+TDM pad)
  __shared__ _Float16 pT[8][16][40];    // per-wave P staging

  const int t  = threadIdx.x;
  const int wv = t >> 5;
  const int l  = t & 31;
  const int lm = l & 15;
  const bool hihalf = (l >= 16);
  const int sel8  = hihalf ? 8  : 0;
  const int sel16 = hihalf ? 16 : 0;

  const int qbase = blockIdx.x * 128;
  const int bh    = blockIdx.y;
  const int bb    = bh >> 4;
  const int hh    = bh & 15;
  const int qrow0 = qbase + wv * 16;
  const unsigned char* mrowp = amask + (size_t)bb * S_;

  const unsigned long long kG = (unsigned long long)(Kh + (size_t)bh * S_ * D_);
  const unsigned long long vG = (unsigned long long)(VT + (size_t)bh * D_ * S_);

  // Q A-fragments resident in VGPRs (D split 0..31 / 32..63)
  const _Float16* qrow = Qh + ((size_t)bh * S_ + qrow0 + lm) * D_;
  v16h aq0 = ld2x8(qrow + sel16,      qrow + sel16 + 8);
  v16h aq1 = ld2x8(qrow + 32 + sel16, qrow + 32 + sel16 + 8);

  v8f oacc[4];
  for (int f = 0; f < 4; ++f) oacc[f] = (v8f){0.f,0.f,0.f,0.f,0.f,0.f,0.f,0.f};
  float mr[8], lr[8];
  for (int r = 0; r < 8; ++r) { mr[r] = -1e30f; lr[r] = 0.f; }

  const int numkt = (qbase + 128) >> 5;

  if (wv == 0) {            // prologue DMA: tile 0 -> buffer 0
    tdm_load_2d(lds_off(&kT[0][0][0]), kG, TDM_W0_K, 64, S_, 64, 32, 64);
    tdm_load_2d(lds_off(&vT[0][0][0]), vG, TDM_W0_V, S_, 64, 32, 64, S_);
  }

  for (int kt = 0; kt < numkt; ++kt) {
    const int kbase = kt * 32;
    const int buf   = kt & 1;
    if (wv == 0) {
      if (kt + 1 < numkt) {           // overlap next tile's DMA with this compute
        const int nb = (kt + 1) & 1, nk = (kt + 1) * 32;
        tdm_load_2d(lds_off(&kT[nb][0][0]), kG + (size_t)nk * D_ * 2,
                    TDM_W0_K, 64, S_, 64, 32, 64);
        tdm_load_2d(lds_off(&vT[nb][0][0]), vG + (size_t)nk * 2,
                    TDM_W0_V, S_, 64, 32, 64, S_);
        __builtin_amdgcn_s_wait_tensorcnt(2);   // current tile (2 loads) complete
      } else {
        __builtin_amdgcn_s_wait_tensorcnt(0);
      }
    }
    __syncthreads();                  // (A) staged tile visible to all waves

    if (kbase <= qrow0 + 15) {        // wave-uniform -> EXEC all-ones for WMMA
      // ---- S = Q K^T (pre-scaled) ----
      v8f c0 = (v8f){0.f,0.f,0.f,0.f,0.f,0.f,0.f,0.f};
      v8f c1 = c0;
      {
        const _Float16* kr0 = &kT[buf][lm][0];
        const _Float16* kr1 = &kT[buf][16 + lm][0];
        v16h b;
        b = ld2x8(kr0 + sel16,      kr0 + sel16 + 8);      c0 = WMMA_F16(aq0, b, c0);
        b = ld2x8(kr0 + 32 + sel16, kr0 + 32 + sel16 + 8); c0 = WMMA_F16(aq1, b, c0);
        b = ld2x8(kr1 + sel16,      kr1 + sel16 + 8);      c1 = WMMA_F16(aq0, b, c1);
        b = ld2x8(kr1 + 32 + sel16, kr1 + 32 + sel16 + 8); c1 = WMMA_F16(aq1, b, c1);
      }
      // ---- causal+pad mask, online softmax (fp32) ----
      const int col0 = kbase + lm, col1 = kbase + 16 + lm;
      const bool mv0 = mrowp[col0] != 0, mv1 = mrowp[col1] != 0;
      for (int r = 0; r < 8; ++r) {
        const int q = qrow0 + r + (hihalf ? 8 : 0);
        float s0v = (mv0 && col0 <= q) ? c0[r] : -1e30f;
        float s1v = (mv1 && col1 <= q) ? c1[r] : -1e30f;
        float tm = fmaxf(s0v, s1v);
        tm = fmaxf(tm, __shfl_xor(tm, 1));
        tm = fmaxf(tm, __shfl_xor(tm, 2));
        tm = fmaxf(tm, __shfl_xor(tm, 4));
        tm = fmaxf(tm, __shfl_xor(tm, 8));
        float mn = fmaxf(mr[r], tm);
        float al = __expf(mr[r] - mn);
        mr[r] = mn;
        float p0 = __expf(s0v - mn);
        float p1 = __expf(s1v - mn);
        float rs = p0 + p1;
        rs += __shfl_xor(rs, 1);
        rs += __shfl_xor(rs, 2);
        rs += __shfl_xor(rs, 4);
        rs += __shfl_xor(rs, 8);
        lr[r] = lr[r] * al + rs;
        oacc[0][r] *= al; oacc[1][r] *= al; oacc[2][r] *= al; oacc[3][r] *= al;
        _Float16* pp = &pT[wv][r + (hihalf ? 8 : 0)][0];
        pp[lm] = (_Float16)p0;
        pp[16 + lm] = (_Float16)p1;
      }
      asm volatile("s_wait_dscnt 0" ::: "memory");  // per-wave LDS RAW on pT
      // ---- O += P V ----
      const _Float16* pr = &pT[wv][lm][0];
      v16h ap = ld2x8(pr + sel16, pr + sel16 + 8);
      for (int f = 0; f < 4; ++f) {
        const _Float16* vrow = &vT[buf][f * 16 + lm][0];
        v16h bvf = ld2x8(vrow + sel16, vrow + sel16 + 8);
        oacc[f] = WMMA_F16(ap, bvf, oacc[f]);
      }
    }
    __syncthreads();                  // (B) buffer drained before reuse at kt+2
  }

  for (int r = 0; r < 8; ++r) {
    const float rl = (lr[r] > 0.f) ? (1.0f / lr[r]) : 0.f;
    const int s = qrow0 + r + (hihalf ? 8 : 0);
    _Float16* dst = Oh + ((size_t)bb * S_ + s) * E_ + hh * D_;
    for (int f = 0; f < 4; ++f) dst[f * 16 + lm] = (_Float16)(oacc[f][r] * rl);
  }
}

// =====================================================================
// Kernel 3: output projection.  out = O @ Wp^T  (fp32 result to d_out)
// A tile (f16) staged via GLOBAL_LOAD_ASYNC_TO_LDS_B128 (ASYNCcnt).
// =====================================================================
__global__ __launch_bounds__(256) void out_proj_kernel(
    const _Float16* __restrict__ Oh, const float* __restrict__ Wp,
    float* __restrict__ out)
{
  __shared__ _Float16 aT[128][40];
  __shared__ _Float16 wT[64][40];

  const int t  = threadIdx.x;
  const int wv = t >> 5;
  const int l  = t & 31;
  const int lm = l & 15;
  const bool hihalf = (l >= 16);
  const int sel8  = hihalf ? 8  : 0;
  const int sel16 = hihalf ? 16 : 0;

  const int ntile = blockIdx.x * 64;
  const int mtile = blockIdx.y * 128;

  v8f acc[4];
  for (int f = 0; f < 4; ++f) acc[f] = (v8f){0.f,0.f,0.f,0.f,0.f,0.f,0.f,0.f};

  for (int kt = 0; kt < E_ / 32; ++kt) {
    __syncthreads();
    // A tile 128x32 f16: async global->LDS copy, no VGPR round-trip
    for (int i = 0; i < 2; ++i) {
      int e = t + 256 * i;
      int row = e >> 2, c8 = (e & 3) * 8;
      unsigned ldst = lds_off(&aT[row][c8]);
      const _Float16* src = Oh + (size_t)(mtile + row) * E_ + kt * 32 + c8;
      asm volatile("global_load_async_to_lds_b128 %0, %1, off"
                   :: "v"(ldst), "v"(src) : "memory");
    }
    for (int i = 0; i < 2; ++i) {              // Wp fp32 -> f16 (needs conversion)
      int e = t + 256 * i;
      int row = e >> 3, c4 = (e & 7) * 4;
      float4 f4 = *(const float4*)(Wp + (size_t)(ntile + row) * E_ + kt * 32 + c4);
      v4h h; h[0]=(_Float16)f4.x; h[1]=(_Float16)f4.y; h[2]=(_Float16)f4.z; h[3]=(_Float16)f4.w;
      *(v4h*)&wT[row][c4] = h;
    }
    asm volatile("s_wait_asynccnt 0x0" ::: "memory");
    __syncthreads();

    const _Float16* arow = &aT[wv * 16 + lm][0];
    v16h a = ld2x8(arow + sel16, arow + sel16 + 8);
    for (int f = 0; f < 4; ++f) {
      const _Float16* brow = &wT[f * 16 + lm][0];
      v16h bf = ld2x8(brow + sel16, brow + sel16 + 8);
      acc[f] = WMMA_F16(a, bf, acc[f]);
    }
  }

  const int mbase = mtile + wv * 16 + (hihalf ? 8 : 0);
  for (int f = 0; f < 4; ++f) {
    const int n = ntile + f * 16 + lm;
    for (int r = 0; r < 8; ++r)
      out[(size_t)(mbase + r) * E_ + n] = acc[f][r];
  }
}

// =====================================================================
extern "C" void kernel_launch(void* const* d_in, const int* in_sizes, int n_in,
                              void* d_out, int out_size, void* d_ws, size_t ws_size,
                              hipStream_t stream) {
  const float*         x     = (const float*)d_in[0];
  const unsigned char* amask = (const unsigned char*)d_in[1];   // bool[B,S]
  const float*         Wq    = (const float*)d_in[2];
  const float*         Wk    = (const float*)d_in[3];
  const float*         Wv    = (const float*)d_in[4];
  const float*         Wp    = (const float*)d_in[5];
  float*               out   = (float*)d_out;

  // f16 workspace: Q,K (B,H,S,D) | V^T (B,H,D,S) | O (B,S,E) — 16 MB each
  const size_t qelems = (size_t)Bv_ * S_ * E_;
  _Float16* Qh = (_Float16*)d_ws;
  _Float16* Kh = Qh + qelems;
  _Float16* VT = Kh + qelems;
  _Float16* Oh = VT + qelems;

  dim3 blk(256, 1, 1);
  qkv_proj_kernel<<<dim3(E_ / 64, (Bv_ * S_) / 128, 3), blk, 0, stream>>>(
      x, Wq, Wk, Wv, Qh, Kh, VT);
  attn_kernel<<<dim3(S_ / 128, Bv_ * H_), blk, 0, stream>>>(
      Qh, Kh, VT, amask, Oh);
  out_proj_kernel<<<dim3(E_ / 64, (Bv_ * S_) / 128), blk, 0, stream>>>(
      Oh, Wp, out);
}
